// ContextNet_34986803593492
// MI455X (gfx1250) — matrix operand order repaired and checked by reading, not compiled, not run on hardware
//
#include <hip/hip_runtime.h>

typedef __attribute__((ext_vector_type(16))) _Float16 v16h;
typedef __attribute__((ext_vector_type(8)))  _Float16 v8h;
typedef __attribute__((ext_vector_type(8)))  float    v8f;

// 16-byte-aligned views (paired/single b128 accesses)
struct __attribute__((packed, aligned(16))) B16 { v16h v; };
struct __attribute__((packed, aligned(16))) S8  { v8h  v; };

static __device__ __forceinline__ long gtid() {
  return (long)blockIdx.x * blockDim.x + threadIdx.x;
}

// ---------------------------------------------------------------- utilities
__global__ void k_zero_f32(float* __restrict__ p, long n) {
  long i = gtid();
  if (i < n) p[i] = 0.0f;
}

// img NCHW f32 [B,3,512,512] -> padded NHWC8 f16 [B,514,514,8]
// (1-pixel zero halo, channels 3..7 = 0)
__global__ void k_img_to_nhwc8_pad(const float* __restrict__ in, _Float16* __restrict__ out,
                                   int B) {
  long i = gtid();
  const int Wp = 514;
  long total = (long)B * Wp * Wp * 8;
  if (i >= total) return;
  int c = (int)(i & 7); long t = i >> 3;
  int xp = (int)(t % Wp); t /= Wp;
  int yp = (int)(t % Wp); int n = (int)(t / Wp);
  _Float16 v = (_Float16)0.0f;
  int x = xp - 1, y = yp - 1;
  if (c < 3 && (unsigned)x < 512u && (unsigned)y < 512u)
    v = (_Float16)in[(((long)n * 3 + c) * 512 + y) * 512 + x];
  out[i] = v;
}

// 2x2 average pool * 0.5  == bilinear half-downsample of flow, rescaled
__global__ void k_halve_flow(const float* __restrict__ in, float* __restrict__ out,
                             int NC, int h, int w) {
  long i = gtid();
  long total = (long)NC * h * w;
  if (i >= total) return;
  int x = (int)(i % w); long t = i / w;
  int y = (int)(t % h); int nc = (int)(t / h);
  const float* ip = in + (((long)nc * (2 * h) + 2 * y) * (2 * w)) + 2 * x;
  out[i] = (ip[0] + ip[1] + ip[2 * w] + ip[2 * w + 1]) * 0.125f;
}

// Pre-swizzle OIHW f32 weights into WMMA A fragments with pos-major K order:
// K = pos*(cpp*32) + channelSlot. frag[((tm*nChunks + chunk)*32 + lane)*16 + e]
// lane<16: M=lane, Kloc in {0..15}; lane>=16: M=lane-16, Kloc in {16..31}.
__global__ void k_pack_wfrag(const float* __restrict__ w, _Float16* __restrict__ frag,
                             int Cin, int cpp, int tilesM) {
  int nChunks = 9 * cpp;
  long i = gtid();
  long total = (long)tilesM * nChunks * 512;
  if (i >= total) return;
  int e    = (int)(i & 15);
  int lane = (int)((i >> 4) & 31);
  long t   = i >> 9;
  int chunk = (int)(t % nChunks);
  int tm    = (int)(t / nChunks);
  int m  = lane & 15, kh = lane >> 4;
  int pos = chunk / cpp, cc = chunk % cpp;
  int slot = cc * 32 + kh * 16 + e;
  _Float16 v = (_Float16)0.0f;
  if (slot < Cin) v = (_Float16)w[((long)(tm * 16 + m) * Cin + slot) * 9 + pos];
  frag[i] = v;
}

// -------------------------------------------------- implicit-GEMM WMMA conv
// Padded NHWC f16 in/out (1-pixel zero halo). One wave computes a
// 16(Cout) x 16(spatial-x) tile at fixed (n, yo). K = 9 positions x CPP*32
// channel slots. No bounds checks: every halo read is in-bounds zero data.
template <int CPP>
__global__ __launch_bounds__(256)
void k_wmma_conv_prelu(const _Float16* __restrict__ in, const _Float16* __restrict__ wfrag,
                       const float* __restrict__ bias, const float* __restrict__ alpha,
                       _Float16* __restrict__ out,
                       int N, int H, int W,        // unpadded input dims
                       int Ho, int Wo, int Cout,   // unpadded output dims
                       int stride, int pixStride) {
  int wave = blockIdx.x * (blockDim.x >> 5) + ((int)threadIdx.x >> 5);
  int lane = (int)threadIdx.x & 31;
  int wT = Wo >> 4, cT = Cout >> 4;
  long tiles = (long)N * cT * Ho * wT;
  if (wave >= tiles) return;            // wave-uniform: EXEC all-ones at WMMA
  int xt = (int)(wave % wT); long t = wave / wT;
  int yo = (int)(t % Ho); t /= Ho;
  int ct = (int)(t % cT); int n = (int)(t / cT);

  int nn = lane & 15, kh = lane >> 4;
  const int Hp = H + 2, Wp = W + 2;
  // padded coords of the (ky=0,kx=0) tap: (yo*stride, ox*stride)
  int xin = ((xt << 4) + nn) * stride;
  int yin = yo * stride;
  const _Float16* base =
      in + (((long)n * Hp + yin) * Wp + xin) * pixStride + (long)kh * 16;

  const int nChunks = 9 * CPP;
  v8f acc = {};
  const v16h* afr = ((const v16h*)wfrag) + (long)ct * nChunks * 32 + lane;

#pragma unroll
  for (int ky = 0; ky < 3; ++ky) {
#pragma unroll
    for (int kx = 0; kx < 3; ++kx) {
      const _Float16* bp = base + ((long)ky * Wp + kx) * pixStride;
#pragma unroll
      for (int cc = 0; cc < CPP; ++cc) {
        v16h a = afr[(long)((ky * 3 + kx) * CPP + cc) * 32];   // immediate offsets
        v16h bv = ((const B16*)(bp + cc * 32))->v;             // 32B, 16B-aligned
        acc = __builtin_amdgcn_wmma_f32_16x16x32_f16(false, a, false, bv,
                                                     (short)0, acc, false, false);
      }
    }
  }

  // D: VGPR r holds M = kh*8 + r -> 8 contiguous NHWC channels per lane.
  // Write the interior of the padded output map.
  const int HoP = Ho + 2, WoP = Wo + 2;
  _Float16* op = out + (((long)n * HoP + yo + 1) * WoP + (xt << 4) + nn + 1) * Cout
                     + (ct << 4) + (kh << 3);
  int cbase = (ct << 4) + (kh << 3);
  v8h o;
#pragma unroll
  for (int r = 0; r < 8; ++r) {
    int co = cbase + r;
    float v = acc[r] + bias[co];
    v = v >= 0.0f ? v : alpha[co] * v;
    o[r] = (_Float16)v;
  }
  ((S8*)op)->v = o;                                            // single 16B store
}

// ------------------------------------------------------- forward splat (avg)
// feat: padded NHWC f16; num: NCHW f32 region of d_out; cnt: [N,h,w]
__global__ void k_splat_scatter(const _Float16* __restrict__ feat, const float* __restrict__ flow,
                                float* __restrict__ num, float* __restrict__ cnt,
                                int N, int C, int h, int w) {
  long i = gtid();
  long hw = (long)h * w;
  if (i >= (long)N * hw) return;
  int x = (int)(i % w); long t = i / w;
  int y = (int)(t % h); int n = (int)(t / h);
  const float* fl = flow + (long)n * 2 * hw;
  float fx = fl[(long)y * w + x] + (float)x;
  float fy = fl[hw + (long)y * w + x] + (float)y;
  float x0f = floorf(fx), y0f = floorf(fy);
  int x0 = (int)x0f, y0 = (int)y0f;
  float ax = fx - x0f, ay = fy - y0f;
  float wt[4] = {(1.f - ax) * (1.f - ay), ax * (1.f - ay), (1.f - ax) * ay, ax * ay};
  int xs[4] = {x0, x0 + 1, x0, x0 + 1};
  int ys[4] = {y0, y0, y0 + 1, y0 + 1};
  bool ok[4]; long idx[4];
#pragma unroll
  for (int q = 0; q < 4; ++q) {
    ok[q] = (unsigned)xs[q] < (unsigned)w && (unsigned)ys[q] < (unsigned)h && wt[q] != 0.0f;
    idx[q] = ok[q] ? ((long)ys[q] * w + xs[q]) : 0;
  }
  float* cntN = cnt + (long)n * hw;
#pragma unroll
  for (int q = 0; q < 4; ++q)
    if (ok[q]) atomicAdd(&cntN[idx[q]], wt[q]);

  const _Float16* fp = feat + (((long)n * (h + 2) + y + 1) * (w + 2) + x + 1) * C;
  float* numN = num + (long)n * C * hw;
  for (int c0 = 0; c0 < C; c0 += 8) {
    v8h f = ((const S8*)(fp + c0))->v;                 // 16B-aligned vector load
#pragma unroll
    for (int r = 0; r < 8; ++r) {
      float fv = (float)f[r];
      float* dst = numN + (long)(c0 + r) * hw;
#pragma unroll
      for (int q = 0; q < 4; ++q)
        if (ok[q]) atomicAdd(dst + idx[q], fv * wt[q]);
    }
  }
}

__global__ void k_splat_div(float* __restrict__ num, const float* __restrict__ cnt,
                            int N, int C, long hw) {
  long i = gtid();
  long total = (long)N * C * hw;
  if (i >= total) return;
  long pix = i % hw; long t = i / hw;
  int n = (int)(t / C);
  float d = cnt[(long)n * hw + pix];
  num[i] = num[i] / (d == 0.0f ? 1.0f : d);
}

// ------------------------------------------------------------------- driver
extern "C" void kernel_launch(void* const* d_in, const int* in_sizes, int n_in,
                              void* d_out, int out_size, void* d_ws, size_t ws_size,
                              hipStream_t stream) {
  (void)in_sizes; (void)n_in; (void)ws_size;
  const int B = 16;
  const float* img  = (const float*)d_in[0];
  const float* flow = (const float*)d_in[1];

  char* wsp = (char*)d_ws;
  auto alloc = [&](size_t bytes) -> char* {
    char* p = wsp;
    wsp += (bytes + 255) & ~(size_t)255;
    return p;
  };
  const size_t imgPadded = (size_t)B * 514 * 514 * 8;    // 33.8M halves (67.6 MB)
  const size_t maxFeatP  = (size_t)B * 258 * 258 * 16;   // 17.0M halves (34.1 MB)
  _Float16* bufX = (_Float16*)alloc(imgPadded * 2 + 256);
  _Float16* bufA = (_Float16*)alloc(maxFeatP * 2 + 256);
  _Float16* bufB = (_Float16*)alloc(maxFeatP * 2 + 256);
  float* flowA = (float*)alloc((size_t)B * 2 * 256 * 256 * 4);
  float* flowB = (float*)alloc((size_t)B * 2 * 256 * 256 * 4);
  float* cnt   = (float*)alloc((size_t)B * 256 * 256 * 4);

  static const int CinA[4]  = {3, 16, 32, 64};
  static const int CoutA[4] = {16, 32, 64, 128};
  static const int cppA[4]  = {1, 1, 1, 2};   // ceil(Cin/32), zero-padded
  static const int cppB[4]  = {1, 1, 2, 4};   // conv-b Cin = Cout
  _Float16* wfragA[4]; _Float16* wfragB[4];
  for (int i = 0; i < 4; ++i) {
    wfragA[i] = (_Float16*)alloc((size_t)(CoutA[i] / 16) * 9 * cppA[i] * 512 * 2);
    wfragB[i] = (_Float16*)alloc((size_t)(CoutA[i] / 16) * 9 * cppB[i] * 512 * 2);
  }

  auto nblk = [](long n, long per) { return dim3((unsigned)((n + per - 1) / per)); };
  auto zero_halves = [&](void* p, long halves) {       // halves is always even
    k_zero_f32<<<nblk(halves / 2, 256), 256, 0, stream>>>((float*)p, halves / 2);
  };
  auto launch_conv = [&](int cpp, const _Float16* in, const _Float16* wf,
                         const float* b, const float* a, _Float16* o,
                         int H, int W, int Ho, int Wo, int Co, int stride, int pixStride) {
    long tiles = (long)B * (Co / 16) * Ho * (Wo / 16);
    dim3 g = nblk(tiles, 8);                           // 8 waves / 256-thread block
    switch (cpp) {
      case 1: k_wmma_conv_prelu<1><<<g, 256, 0, stream>>>(in, wf, b, a, o, B, H, W, Ho, Wo, Co, stride, pixStride); break;
      case 2: k_wmma_conv_prelu<2><<<g, 256, 0, stream>>>(in, wf, b, a, o, B, H, W, Ho, Wo, Co, stride, pixStride); break;
      default: k_wmma_conv_prelu<4><<<g, 256, 0, stream>>>(in, wf, b, a, o, B, H, W, Ho, Wo, Co, stride, pixStride); break;
    }
  };

  // d_out is the splat accumulator: zero every call (deterministic replay)
  k_zero_f32<<<nblk(out_size, 256), 256, 0, stream>>>((float*)d_out, (long)out_size);
  k_img_to_nhwc8_pad<<<nblk((long)imgPadded, 256), 256, 0, stream>>>(img, bufX, B);

  const long outOff[4] = {0, 16777216, 25165824, 29360128};
  const float* curFlow = flow;
  _Float16* cin = bufX;
  int Hin = 512, Win = 512, pixIn = 8;   // img padded to 8 channels/pixel
  for (int blk = 0; blk < 4; ++blk) {
    int Ci = CinA[blk], Co = CoutA[blk];
    int Ho = Hin >> 1, Wo = Win >> 1;
    const float* wa = (const float*)d_in[2 + blk * 6 + 0];
    const float* ba = (const float*)d_in[2 + blk * 6 + 1];
    const float* aa = (const float*)d_in[2 + blk * 6 + 2];
    const float* wb = (const float*)d_in[2 + blk * 6 + 3];
    const float* bb = (const float*)d_in[2 + blk * 6 + 4];
    const float* ab = (const float*)d_in[2 + blk * 6 + 5];

    long pa = (long)(Co / 16) * 9 * cppA[blk] * 512;
    k_pack_wfrag<<<nblk(pa, 256), 256, 0, stream>>>(wa, wfragA[blk], Ci, cppA[blk], Co / 16);
    long pb = (long)(Co / 16) * 9 * cppB[blk] * 512;
    k_pack_wfrag<<<nblk(pb, 256), 256, 0, stream>>>(wb, wfragB[blk], Co, cppB[blk], Co / 16);

    long padOutHalves = (long)B * (Ho + 2) * (Wo + 2) * Co;
    // conv a: stride 2 (zero the padded region first so the halo is 0)
    zero_halves(bufA, padOutHalves);
    launch_conv(cppA[blk], cin, wfragA[blk], ba, aa, bufA,
                Hin, Win, Ho, Wo, Co, 2, pixIn);
    // conv b: stride 1
    _Float16* outb = (cin == bufX) ? bufB : bufX;
    zero_halves(outb, padOutHalves);
    launch_conv(cppB[blk], bufA, wfragB[blk], bb, ab, outb,
                Ho, Wo, Ho, Wo, Co, 1, Co);

    // halve flow
    float* fdst = (blk & 1) ? flowB : flowA;
    long fN = (long)B * 2 * Ho * Wo;
    k_halve_flow<<<nblk(fN, 256), 256, 0, stream>>>(curFlow, fdst, B * 2, Ho, Wo);
    curFlow = fdst;

    // softsplat average
    long pix = (long)B * Ho * Wo;
    k_zero_f32<<<nblk(pix, 256), 256, 0, stream>>>(cnt, pix);
    float* num = (float*)d_out + outOff[blk];
    k_splat_scatter<<<nblk(pix, 256), 256, 0, stream>>>(outb, fdst, num, cnt, B, Co, Ho, Wo);
    long tot = (long)B * Co * Ho * Wo;
    k_splat_div<<<nblk(tot, 256), 256, 0, stream>>>(num, cnt, B, Co, (long)Ho * Wo);

    cin = outb; Hin = Ho; Win = Wo; pixIn = Co;
  }
}
